// LFQQuantizerEMA_KMeans_31052613550670
// MI455X (gfx1250) — compile-verified, matrix operand order checked by reading, not run on hardware
//
#include <hip/hip_runtime.h>

// ---------------------------------------------------------------------------
// VQ nearest-codebook lookup for MI455X (gfx1250, wave32, WMMA f16 16x16x32).
//   z_e:      [65536, 256] f32
//   codebook: [1024, 256]  f32
//   out: z_q [65536,256] f32  ++  indices [65536] int32 (concatenated in d_out)
// argmin_k ( ||c_k||^2 - 2 z.c_k )   (||z||^2 constant per row -> dropped)
// B (codebook) tiles stream global->LDS via GLOBAL_LOAD_ASYNC_TO_LDS_B128
// (ASYNCcnt), so no staging VGPRs and no spills; A tiles live in VGPRs.
// ---------------------------------------------------------------------------

typedef __attribute__((ext_vector_type(16))) _Float16 v16h;
typedef __attribute__((ext_vector_type(8)))  float    v8f;

// async-to-LDS builtin wants pointers to 16B int vectors in AS1 (global) / AS3 (LDS)
typedef int v4i_b128 __attribute__((vector_size(16)));
typedef __attribute__((address_space(1))) v4i_b128 as1_v4i;
typedef __attribute__((address_space(3))) v4i_b128 as3_v4i;

#define NUM_CODES 1024
#define CODE_DIM  256
#define BATCH     65536

#define NTILES       (NUM_CODES / 16)   // 64 N-tiles of 16 codes
#define KSTEPS       (CODE_DIM / 32)    // 8 WMMA K-steps of 32
#define FRAG_HALVES  512                // halves per (ntile,kstep) B fragment
#define CH_NT        2                  // ntiles per LDS chunk
#define CHUNK_HALVES (CH_NT * KSTEPS * FRAG_HALVES)  // 8192 halves = 16 KB
#define NCHUNK       (NTILES / CH_NT)                // 32
#define CHUNK_BYTES  (CHUNK_HALVES * 2)              // 16384
#define B128_PT      (CHUNK_BYTES / (256 * 16))     // 4 async b128 per thread

#if __has_builtin(__builtin_amdgcn_global_load_async_to_lds_b128)
#define HAVE_ASYNC_LDS 1
#else
#define HAVE_ASYNC_LDS 0
#endif

__device__ __forceinline__ void wait_async0() {
#if HAVE_ASYNC_LDS
#if __has_builtin(__builtin_amdgcn_s_wait_asynccnt)
    __builtin_amdgcn_s_wait_asynccnt(0);
#else
    asm volatile("s_wait_asynccnt 0x0" ::: "memory");
#endif
#endif
}

// Stage one 16 KB codebook-fragment chunk into an LDS buffer.
__device__ __forceinline__ void stage_chunk(const _Float16* __restrict__ frag,
                                            _Float16* ldsDst, int chunk, int tid) {
#if HAVE_ASYNC_LDS
    const char* g = (const char*)frag + (size_t)chunk * CHUNK_BYTES + tid * 16;
    char*       l = (char*)ldsDst + tid * 16;
    #pragma unroll
    for (int i = 0; i < B128_PT; ++i)
        __builtin_amdgcn_global_load_async_to_lds_b128(
            (as1_v4i*)(g + i * 4096),
            (as3_v4i*)(l + i * 4096),
            /*offset=*/0, /*cpol=*/0);
#else
    const uint4* g = (const uint4*)frag + (size_t)chunk * (CHUNK_BYTES / 16);
    uint4*       l = (uint4*)ldsDst;
    #pragma unroll
    for (int i = 0; i < B128_PT; ++i) l[i * 256 + tid] = g[i * 256 + tid];
#endif
}

// --- WMMA B-fragment index map (32x16 f16 B operand, per ISA 16-bit layouts):
// lane l holds column n = l&15;  hi = l>>4 adds +8 to K;  halves j=0..15:
//   g = j>>1, p = j&1 :  k = (g<4) ? 2g+p+8*hi : 16+2*(g-4)+p+8*hi
// Fragments stored linearly: frag[((ntile*8+ks)*32 + lane)*16 + j]

__global__ void prep_frags(const float* __restrict__ cb, _Float16* __restrict__ frag) {
    int t = blockIdx.x * 256 + threadIdx.x;           // one f16 element per thread
    int j     = t & 15;
    int lane  = (t >> 4) & 31;
    int ks    = (t >> 9) & 7;
    int ntile = t >> 12;
    int hi = lane >> 4;
    int nl = lane & 15;
    int g = j >> 1, p = j & 1;
    int k_local = (g < 4) ? (2 * g + p + 8 * hi) : (16 + 2 * (g - 4) + p + 8 * hi);
    int code = ntile * 16 + nl;
    int d    = ks * 32 + k_local;
    frag[t] = (_Float16)cb[code * CODE_DIM + d];      // coalesced 2B writes, gathered reads (L2)
}

__global__ void prep_norms(const float* __restrict__ cb, float* __restrict__ norms) {
    int lane = threadIdx.x & 31;
    int wv   = threadIdx.x >> 5;
    int code = blockIdx.x * 8 + wv;                   // one wave per code
    const float* row = cb + (size_t)code * CODE_DIM;
    float s = 0.f;
    #pragma unroll
    for (int i = 0; i < CODE_DIM / 32; ++i) {
        float v = row[lane + 32 * i];
        s += v * v;
    }
    #pragma unroll
    for (int off = 16; off >= 1; off >>= 1) s += __shfl_xor(s, off, 32);
    if (lane == 0) norms[code] = s;
}

__launch_bounds__(256)
__global__ void vq_argmin_gather(const float*    __restrict__ z,
                                 const float*    __restrict__ cb,
                                 const _Float16* __restrict__ frag,
                                 const float*    __restrict__ norms,
                                 float*          __restrict__ zq,
                                 int*            __restrict__ idxOut) {
    __shared__ __align__(32) _Float16 ldsB[2][CHUNK_HALVES];   // 32 KB double-buffered B tiles
    __shared__ float ldsNorm[NUM_CODES];                       // 4 KB
    __shared__ int   ldsIdx[256];                              // 1 KB

    const int tid  = threadIdx.x;
    const int lane = tid & 31;
    const int wv   = tid >> 5;           // wave 0..7
    const int hi   = lane >> 4;          // half-wave select
    const int nl   = lane & 15;

    for (int i = tid; i < NUM_CODES; i += 256) ldsNorm[i] = norms[i];

    // ---- A fragments: 32 rows per wave (2 subtiles of 16), f32 -> f16, kept in VGPRs.
    // 16-bit A 16x32 layout: lane holds row m = l&15; halves j: k = (j>>3)*16 + hi*8 + (j&7)
    const int rowBase = blockIdx.x * 256 + wv * 32;
    v16h Af[2][8];
    #pragma unroll
    for (int s = 0; s < 2; ++s) {
        const float* zr = z + (size_t)(rowBase + s * 16 + nl) * CODE_DIM;
        #pragma unroll
        for (int ks = 0; ks < 8; ++ks) {
            int k0 = ks * 32 + hi * 8;
            float4 a0 = *(const float4*)(zr + k0);
            float4 a1 = *(const float4*)(zr + k0 + 4);
            float4 a2 = *(const float4*)(zr + k0 + 16);
            float4 a3 = *(const float4*)(zr + k0 + 20);
            v16h A;
            A[0]  = (_Float16)a0.x; A[1]  = (_Float16)a0.y; A[2]  = (_Float16)a0.z; A[3]  = (_Float16)a0.w;
            A[4]  = (_Float16)a1.x; A[5]  = (_Float16)a1.y; A[6]  = (_Float16)a1.z; A[7]  = (_Float16)a1.w;
            A[8]  = (_Float16)a2.x; A[9]  = (_Float16)a2.y; A[10] = (_Float16)a2.z; A[11] = (_Float16)a2.w;
            A[12] = (_Float16)a3.x; A[13] = (_Float16)a3.y; A[14] = (_Float16)a3.z; A[15] = (_Float16)a3.w;
            Af[s][ks] = A;
        }
    }

    float minv[2][8];
    int   mini[2][8];
    #pragma unroll
    for (int s = 0; s < 2; ++s)
        #pragma unroll
        for (int r = 0; r < 8; ++r) { minv[s][r] = 3.4e38f; mini[s][r] = 0; }

    // ---- K sweep over 64 N-tiles, async-LDS double-buffered B chunks -------
    stage_chunk(frag, ldsB[0], 0, tid);
    wait_async0();
    __syncthreads();                                  // chunk 0 + ldsNorm visible

    for (int c = 0; c < NCHUNK; ++c) {
        const int buf = c & 1;
        if (c + 1 < NCHUNK)                           // async prefetch next chunk
            stage_chunk(frag, ldsB[buf ^ 1], c + 1, tid);

        #pragma unroll
        for (int nt = 0; nt < CH_NT; ++nt) {
            const int code = (c * CH_NT + nt) * 16 + nl;   // this lane's N column
            const float cn = ldsNorm[code];
            v8f acc0 = {}, acc1 = {};
            const _Float16* bb = &ldsB[buf][nt * KSTEPS * FRAG_HALVES + lane * 16];
            v16h Bcur = *(const v16h*)bb;                  // pipeline B one K-step ahead
            #pragma unroll
            for (int ks = 0; ks < 8; ++ks) {
                v16h Bnxt = Bcur;
                if (ks < 7) Bnxt = *(const v16h*)(bb + (ks + 1) * FRAG_HALVES);
                acc0 = __builtin_amdgcn_wmma_f32_16x16x32_f16(false, Af[0][ks], false, Bcur,
                                                              (short)0, acc0, false, false);
                acc1 = __builtin_amdgcn_wmma_f32_16x16x32_f16(false, Af[1][ks], false, Bcur,
                                                              (short)0, acc1, false, false);
                Bcur = Bnxt;
            }
            #pragma unroll
            for (int r = 0; r < 8; ++r) {             // C/D layout: lane=N, VGPR r = M row
                float v0 = cn - 2.0f * acc0[r];
                if (v0 < minv[0][r]) { minv[0][r] = v0; mini[0][r] = code; }
                float v1 = cn - 2.0f * acc1[r];
                if (v1 < minv[1][r]) { minv[1][r] = v1; mini[1][r] = code; }
            }
        }
        wait_async0();                                // next chunk fully in LDS
        __syncthreads();                              // all waves done with buf
    }

    // ---- cross-lane argmin within each 16-lane N group (first-index ties) --
    #pragma unroll
    for (int s = 0; s < 2; ++s) {
        #pragma unroll
        for (int r = 0; r < 8; ++r) {
            float v  = minv[s][r];
            int   ix = mini[s][r];
            #pragma unroll
            for (int off = 8; off >= 1; off >>= 1) {
                float ov = __shfl_xor(v, off, 16);
                int   oi = __shfl_xor(ix, off, 16);
                if (ov < v || (ov == v && oi < ix)) { v = ov; ix = oi; }
            }
            if (nl == 0) {                            // lane 0 -> rows 0..7, lane 16 -> rows 8..15
                int rowLocal = wv * 32 + s * 16 + r + 8 * hi;
                ldsIdx[rowLocal] = ix;
                idxOut[blockIdx.x * 256 + rowLocal] = ix;
            }
        }
    }
    __syncthreads();

    // ---- gather z_q = codebook[idx] (fp32, coalesced: wave per row) --------
    for (int rr = 0; rr < 32; ++rr) {
        const int rowLocal = wv * 32 + rr;
        const int code = ldsIdx[rowLocal];
        const float4* src = (const float4*)(cb + (size_t)code * CODE_DIM);
        float4* dst = (float4*)(zq + (size_t)(blockIdx.x * 256 + rowLocal) * CODE_DIM);
        dst[lane * 2]     = src[lane * 2];
        dst[lane * 2 + 1] = src[lane * 2 + 1];
    }
}

extern "C" void kernel_launch(void* const* d_in, const int* in_sizes, int n_in,
                              void* d_out, int out_size, void* d_ws, size_t ws_size,
                              hipStream_t stream) {
    (void)in_sizes; (void)n_in; (void)out_size; (void)ws_size;
    const float* z  = (const float*)d_in[0];
    const float* cb = (const float*)d_in[1];

    _Float16* frag  = (_Float16*)d_ws;                                   // 512 KB
    float*    norms = (float*)((char*)d_ws +
                      (size_t)NTILES * KSTEPS * FRAG_HALVES * sizeof(_Float16));  // +4 KB

    float* zq     = (float*)d_out;
    int*   idxOut = (int*)(zq + (size_t)BATCH * CODE_DIM);

    prep_frags<<<(NTILES * KSTEPS * FRAG_HALVES) / 256, 256, 0, stream>>>(cb, frag);
    prep_norms<<<NUM_CODES / 8, 256, 0, stream>>>(cb, norms);
    vq_argmin_gather<<<BATCH / 256, 256, 0, stream>>>(z, cb, frag, norms, zq, idxOut);
}
